// WienerSolver_23665269800997
// MI455X (gfx1250) — matrix operand order repaired and checked by reading, not compiled, not run on hardware
//
#include <hip/hip_runtime.h>
#include <math.h>

// Wiener deconvolution via DFT-as-matmul on CDNA5 using BF16x3 FP32 emulation.
//
//  x = irfft2( conj(H)/(|H|^2 + 0.1*L^2) * rfft2(y) ), clip [-1,1]
//
// 2D DFT = F*Y*F with symmetric F = A + iB (A=cos, B=-sin). Each stage is
// "right-multiply by the symmetric DFT matrix, store transposed", so the DFT
// matrix is always the WMMA B-operand (loaded row-wise via symmetry) and the
// image operand is always a row-major A-operand.
//
// Precision path: every f32 value is split x = hi + lo (bf16 each); products
// are computed as hi*hi + hi*lo + lo*hi with V_WMMA_F32_16X16X32_BF16 and f32
// accumulation (error ~2^-18 relative). This runs ~2.7x faster than the
// V_WMMA_F32_16X16X4_F32 path (8x MACs per instruction, 3 instrs per product).
// The problem is compute bound: ~155 GFLOP of matmul vs ~1.5 GB of HBM traffic
// (~65 us at 23.3 TB/s); DFT matrices + per-stage tiles are L2-resident.

typedef __attribute__((ext_vector_type(16))) __bf16 v16bf;
typedef __attribute__((ext_vector_type(8)))  float v8f;

#define KN 512
#define NNE (KN * KN)
#define IMGS 48
#define PSFK 15
#define TWO_PI 6.28318530717958647692f

union Frag {
    v16bf v;
    uint4 q[2];
};

__device__ __forceinline__ unsigned short bf_bits(__bf16 x) {
    return __builtin_bit_cast(unsigned short, x);
}
__device__ __forceinline__ void splitf(float x, __bf16& h, __bf16& l) {
    __bf16 hb = (__bf16)x;          // RNE
    h = hb;
    l = (__bf16)(x - (float)hb);
}

// ---------------------------------------------------------------------------
// DFT matrices, pre-split to bf16 hi/lo: Am = cos(-2pi*k*n/N), Bm = sin(...),
// Bn = -Bm (needed so dual GEMMs are always pure adds: bf16 WMMA has no A/B neg)
// ---------------------------------------------------------------------------
__global__ void build_dft_split_kernel(__bf16* __restrict__ AmH, __bf16* __restrict__ AmL,
                                       __bf16* __restrict__ BmH, __bf16* __restrict__ BmL,
                                       __bf16* __restrict__ BnH, __bf16* __restrict__ BnL) {
    int idx = blockIdx.x * blockDim.x + threadIdx.x;
    if (idx >= KN * KN) return;
    int k = idx >> 9;
    int n = idx & (KN - 1);
    int m = (k * n) & (KN - 1);                 // exact (k*n) mod 512
    float theta = -TWO_PI * (float)m / (float)KN;
    float s, c;
    __sincosf(theta, &s, &c);
    __bf16 h, l;
    splitf(c, h, l);  AmH[idx] = h; AmL[idx] = l;
    splitf(s, h, l);  BmH[idx] = h; BmL[idx] = l;
    splitf(-s, h, l); BnH[idx] = h; BnL[idx] = l;
}

// ---------------------------------------------------------------------------
// Split input images y (f32) into bf16 hi/lo planes: layout [img][2][512*512]
// ---------------------------------------------------------------------------
__global__ void split_y_kernel(const float* __restrict__ y, __bf16* __restrict__ yS) {
    int idx = blockIdx.x * blockDim.x + threadIdx.x;
    if (idx >= IMGS * NNE) return;
    int img = idx / NNE;
    int r   = idx - img * NNE;
    __bf16 h, l;
    splitf(y[idx], h, l);
    __bf16* base = yS + (size_t)img * 2 * NNE;
    base[r]       = h;
    base[NNE + r] = l;
}

// ---------------------------------------------------------------------------
// PSF row DFT: G[img][u][k2] = sum_v psf[img][u][v] * e^{-i*2pi*k2*(v-7)/N}
// ---------------------------------------------------------------------------
__global__ void psf_row_dft_kernel(const float* __restrict__ psf,
                                   float2* __restrict__ G) {
    int idx = blockIdx.x * blockDim.x + threadIdx.x;
    if (idx >= IMGS * PSFK * KN) return;
    int k2  = idx & (KN - 1);
    int u   = (idx >> 9) % PSFK;
    int img = idx / (PSFK * KN);

    float a = -TWO_PI * (float)k2 / (float)KN;
    float sr, cr;
    __sincosf(a, &sr, &cr);
    float wr, wi;
    __sincosf(-7.0f * a, &wi, &wr);            // start at v=0 -> angle -7a

    const float* p = psf + (size_t)img * PSFK * PSFK + (size_t)u * PSFK;
    float gre = 0.f, gim = 0.f;
#pragma unroll
    for (int v = 0; v < PSFK; ++v) {
        float pv = p[v];
        gre = fmaf(pv, wr, gre);
        gim = fmaf(pv, wi, gim);
        float nwr = wr * cr - wi * sr;
        wi = wr * sr + wi * cr;
        wr = nwr;
    }
    G[(size_t)(img * PSFK + u) * KN + k2] = make_float2(gre, gim);
}

// ---------------------------------------------------------------------------
// Pointwise Wiener filter (Z natural layout, hi/lo bf16 in, hi/lo bf16 out).
//   H  = sum_u e^{-i*2pi*k1*(u-7)/N} * G[img][u][k2]
//   L  = 4 - 2cos(2pi*k1/N) - 2cos(2pi*k2/N)
//   Zf = conj(H)/(|H|^2 + 0.1*L^2) * Z
// ---------------------------------------------------------------------------
__global__ void wiener_filter_kernel(const float2* __restrict__ G,
                                     const __bf16* __restrict__ ZreP,
                                     const __bf16* __restrict__ ZimP,
                                     __bf16* __restrict__ OreP,
                                     __bf16* __restrict__ OimP) {
    int idx = blockIdx.x * blockDim.x + threadIdx.x;
    if (idx >= IMGS * NNE) return;
    int img = idx / NNE;
    int r   = idx - img * NNE;
    int k2  = r & (KN - 1);
    int k1  = r >> 9;

    float a = -TWO_PI * (float)k1 / (float)KN;
    float sr, cr;
    __sincosf(a, &sr, &cr);                    // cr = cos(2pi*k1/N)
    float wr, wi;
    __sincosf(-7.0f * a, &wi, &wr);

    const float2* g = G + (size_t)img * PSFK * KN + k2;
    float hre = 0.f, him = 0.f;
#pragma unroll
    for (int u = 0; u < PSFK; ++u) {
        float2 gv = g[(size_t)u * KN];
        hre += wr * gv.x - wi * gv.y;
        him += wr * gv.y + wi * gv.x;
        float nwr = wr * cr - wi * sr;
        wi = wr * sr + wi * cr;
        wr = nwr;
    }

    float ck2 = __cosf(TWO_PI * (float)k2 / (float)KN);
    float L = 4.0f - 2.0f * cr - 2.0f * ck2;
    float denom = hre * hre + him * him + 0.1f * L * L;
    float inv = 1.0f / denom;

    const __bf16* zrB = ZreP + (size_t)img * 2 * NNE;
    const __bf16* ziB = ZimP + (size_t)img * 2 * NNE;
    float zr = (float)zrB[r] + (float)zrB[NNE + r];
    float zi = (float)ziB[r] + (float)ziB[NNE + r];

    float ore = (hre * zr + him * zi) * inv;
    float oim = (hre * zi - him * zr) * inv;

    __bf16 h, l;
    __bf16* orB = OreP + (size_t)img * 2 * NNE;
    __bf16* oiB = OimP + (size_t)img * 2 * NNE;
    splitf(ore, h, l); orB[r] = h; orB[NNE + r] = l;
    splitf(oim, h, l); oiB[r] = h; oiB[NNE + r] = l;
}

// ---------------------------------------------------------------------------
// BF16x3 transpose-store GEMM:
//   D^T = P1*Q1 (+ P2*Q2), each product = hi*hi + hi*lo + lo*hi in bf16 WMMA.
// P*: row-major image operand, hi plane at P, lo plane at P+NNE (per image).
// Q*: symmetric 512x512 DFT matrix (hi/lo planes); symmetry lets us load the
//     WMMA B-fragment (column slice) as a contiguous row slice.
// Output: FINAL -> f32, scaled + clipped; else bf16 hi/lo split planes.
// One wave computes a 64x32 tile = 4x2 accumulators; block = 4 waves = 64x128.
// ---------------------------------------------------------------------------
template <bool DUAL, bool FINAL>
__global__ __launch_bounds__(128) void gemm_bf16x3_kernel(
    const __bf16* __restrict__ P1, unsigned long long p1Stride,
    const __bf16* __restrict__ Q1H, const __bf16* __restrict__ Q1L,
    const __bf16* __restrict__ P2, unsigned long long p2Stride,
    const __bf16* __restrict__ Q2H, const __bf16* __restrict__ Q2L,
    float scale, void* __restrict__ Dout, unsigned long long dStride) {

    const int img  = blockIdx.z;
    const int wave = threadIdx.x >> 5;
    const int lane = threadIdx.x & 31;
    const int half = lane >> 4;                // K-chunk select within fragment
    const int lid  = lane & 15;
    const int row0 = blockIdx.y * 64;                  // 8 row tiles
    const int col0 = (blockIdx.x * 4 + wave) * 32;     // 4 blk * 4 waves * 32

    const __bf16* p1h = P1 + (size_t)img * p1Stride;
    const __bf16* p1l = p1h + NNE;
    const __bf16* p2h = DUAL ? P2 + (size_t)img * p2Stride : nullptr;
    const __bf16* p2l = DUAL ? p2h + NNE : nullptr;

    v8f acc[4][2] = {};

    for (int k = 0; k < KN; k += 32) {
        {   // ---- product 1 ----
            Frag aH[4], aL[4];
#pragma unroll
            for (int tr = 0; tr < 4; ++tr) {
                // A-frag: lane holds row (row0+tr*16+lid), K = k+half*8+{0..7}
                // then K = k+16+half*8+{0..7}  (two 16B chunks)
                size_t off = (size_t)(row0 + tr * 16 + lid) * KN + k + half * 8;
                aH[tr].q[0] = *(const uint4*)(p1h + off);
                aH[tr].q[1] = *(const uint4*)(p1h + off + 16);
                aL[tr].q[0] = *(const uint4*)(p1l + off);
                aL[tr].q[1] = *(const uint4*)(p1l + off + 16);
            }
            Frag bH[2], bL[2];
#pragma unroll
            for (int tc = 0; tc < 2; ++tc) {
                // B-frag via symmetry: Q[k',c] = Q[c,k'];
                // lane holds col (col0+tc*16+lid), K = k+half*8+{0..7}, then k+16+half*8+{0..7}
                size_t boff = (size_t)(col0 + tc * 16 + lid) * KN + k + half * 8;
                bH[tc].q[0] = *(const uint4*)(Q1H + boff);
                bH[tc].q[1] = *(const uint4*)(Q1H + boff + 16);
                bL[tc].q[0] = *(const uint4*)(Q1L + boff);
                bL[tc].q[1] = *(const uint4*)(Q1L + boff + 16);
            }
#pragma unroll
            for (int tr = 0; tr < 4; ++tr)
#pragma unroll
                for (int tc = 0; tc < 2; ++tc) {
                    acc[tr][tc] = __builtin_amdgcn_wmma_f32_16x16x32_bf16(
                        false, aH[tr].v, false, bH[tc].v, (short)0, acc[tr][tc], false, false);
                    acc[tr][tc] = __builtin_amdgcn_wmma_f32_16x16x32_bf16(
                        false, aH[tr].v, false, bL[tc].v, (short)0, acc[tr][tc], false, false);
                    acc[tr][tc] = __builtin_amdgcn_wmma_f32_16x16x32_bf16(
                        false, aL[tr].v, false, bH[tc].v, (short)0, acc[tr][tc], false, false);
                }
        }
        if constexpr (DUAL) {   // ---- product 2 (sign folded into Q2 = -B) ----
            Frag aH[4], aL[4];
#pragma unroll
            for (int tr = 0; tr < 4; ++tr) {
                size_t off = (size_t)(row0 + tr * 16 + lid) * KN + k + half * 8;
                aH[tr].q[0] = *(const uint4*)(p2h + off);
                aH[tr].q[1] = *(const uint4*)(p2h + off + 16);
                aL[tr].q[0] = *(const uint4*)(p2l + off);
                aL[tr].q[1] = *(const uint4*)(p2l + off + 16);
            }
            Frag bH[2], bL[2];
#pragma unroll
            for (int tc = 0; tc < 2; ++tc) {
                size_t boff = (size_t)(col0 + tc * 16 + lid) * KN + k + half * 8;
                bH[tc].q[0] = *(const uint4*)(Q2H + boff);
                bH[tc].q[1] = *(const uint4*)(Q2H + boff + 16);
                bL[tc].q[0] = *(const uint4*)(Q2L + boff);
                bL[tc].q[1] = *(const uint4*)(Q2L + boff + 16);
            }
#pragma unroll
            for (int tr = 0; tr < 4; ++tr)
#pragma unroll
                for (int tc = 0; tc < 2; ++tc) {
                    acc[tr][tc] = __builtin_amdgcn_wmma_f32_16x16x32_bf16(
                        false, aH[tr].v, false, bH[tc].v, (short)0, acc[tr][tc], false, false);
                    acc[tr][tc] = __builtin_amdgcn_wmma_f32_16x16x32_bf16(
                        false, aH[tr].v, false, bL[tc].v, (short)0, acc[tr][tc], false, false);
                    acc[tr][tc] = __builtin_amdgcn_wmma_f32_16x16x32_bf16(
                        false, aL[tr].v, false, bH[tc].v, (short)0, acc[tr][tc], false, false);
                }
        }
    }

    // Transposed store: D^T[n][m]; lane's 8 accum elems are contiguous in m.
#pragma unroll
    for (int tr = 0; tr < 4; ++tr) {
#pragma unroll
        for (int tc = 0; tc < 2; ++tc) {
            const int m0 = row0 + tr * 16 + half * 8;
            const int n  = col0 + tc * 16 + lid;
            v8f v = acc[tr][tc];
            if constexpr (FINAL) {
                float* o = (float*)Dout + (size_t)img * dStride + (size_t)n * KN + m0;
                float4 f0, f1;
                f0.x = fminf(fmaxf(v[0] * scale, -1.f), 1.f);
                f0.y = fminf(fmaxf(v[1] * scale, -1.f), 1.f);
                f0.z = fminf(fmaxf(v[2] * scale, -1.f), 1.f);
                f0.w = fminf(fmaxf(v[3] * scale, -1.f), 1.f);
                f1.x = fminf(fmaxf(v[4] * scale, -1.f), 1.f);
                f1.y = fminf(fmaxf(v[5] * scale, -1.f), 1.f);
                f1.z = fminf(fmaxf(v[6] * scale, -1.f), 1.f);
                f1.w = fminf(fmaxf(v[7] * scale, -1.f), 1.f);
                *(float4*)(o)     = f0;
                *(float4*)(o + 4) = f1;
            } else {
                __bf16* oh = (__bf16*)Dout + (size_t)img * dStride;
                __bf16* ol = oh + NNE;
                unsigned int hw[4], lw[4];
#pragma unroll
                for (int e = 0; e < 4; ++e) {
                    __bf16 h0, l0, h1, l1;
                    splitf(v[2 * e], h0, l0);
                    splitf(v[2 * e + 1], h1, l1);
                    hw[e] = (unsigned)bf_bits(h0) | ((unsigned)bf_bits(h1) << 16);
                    lw[e] = (unsigned)bf_bits(l0) | ((unsigned)bf_bits(l1) << 16);
                }
                uint4 hq, lq;
                hq.x = hw[0]; hq.y = hw[1]; hq.z = hw[2]; hq.w = hw[3];
                lq.x = lw[0]; lq.y = lw[1]; lq.z = lw[2]; lq.w = lw[3];
                *(uint4*)(oh + (size_t)n * KN + m0) = hq;
                *(uint4*)(ol + (size_t)n * KN + m0) = lq;
            }
        }
    }
}

// ---------------------------------------------------------------------------
// Host-side orchestration
// ---------------------------------------------------------------------------
extern "C" void kernel_launch(void* const* d_in, const int* in_sizes, int n_in,
                              void* d_out, int out_size, void* d_ws, size_t ws_size,
                              hipStream_t stream) {
    const float* y   = (const float*)d_in[0];   // [16,3,512,512]
    const float* psf = (const float*)d_in[1];   // [16,3,15,15]
    float* out = (float*)d_out;                 // [16,3,512,512]

    // Workspace layout (bytes): ~246 MB total
    char* w = (char*)d_ws;
    __bf16* AmH = (__bf16*)w; w += (size_t)NNE * 2;
    __bf16* AmL = (__bf16*)w; w += (size_t)NNE * 2;
    __bf16* BmH = (__bf16*)w; w += (size_t)NNE * 2;
    __bf16* BmL = (__bf16*)w; w += (size_t)NNE * 2;
    __bf16* BnH = (__bf16*)w; w += (size_t)NNE * 2;
    __bf16* BnL = (__bf16*)w; w += (size_t)NNE * 2;
    float2* G   = (float2*)w; w += (size_t)IMGS * PSFK * KN * sizeof(float2);
    const size_t planePair = (size_t)2 * NNE;           // hi+lo, bf16 elements
    __bf16* ySp  = (__bf16*)w; w += (size_t)IMGS * planePair * 2;
    __bf16* W1re = (__bf16*)w; w += (size_t)IMGS * planePair * 2;
    __bf16* W1im = (__bf16*)w; w += (size_t)IMGS * planePair * 2;
    __bf16* W2re = (__bf16*)w; w += (size_t)IMGS * planePair * 2;
    __bf16* W2im = (__bf16*)w;

    build_dft_split_kernel<<<(KN * KN) / 256, 256, 0, stream>>>(AmH, AmL, BmH, BmL, BnH, BnL);
    split_y_kernel<<<(IMGS * NNE) / 256, 256, 0, stream>>>(y, ySp);
    psf_row_dft_kernel<<<(IMGS * PSFK * KN) / 256, 256, 0, stream>>>(psf, G);

    dim3 gg(4, 8, IMGS);
    const float inv_nn = 1.0f / ((float)KN * (float)KN);

    // G1: W1 = (y * (A + iB))^T       (row-stage of forward FFT, y real)
    gemm_bf16x3_kernel<false, false><<<gg, 128, 0, stream>>>(
        ySp, planePair, AmH, AmL, nullptr, 0ull, nullptr, nullptr, 1.f, W1re, planePair);
    gemm_bf16x3_kernel<false, false><<<gg, 128, 0, stream>>>(
        ySp, planePair, BmH, BmL, nullptr, 0ull, nullptr, nullptr, 1.f, W1im, planePair);
    // G2: Z = (W1 * (A + iB))^T       -> natural orientation after 2 transposes
    gemm_bf16x3_kernel<true, false><<<gg, 128, 0, stream>>>(
        W1re, planePair, AmH, AmL, W1im, planePair, BnH, BnL, 1.f, W2re, planePair);
    gemm_bf16x3_kernel<true, false><<<gg, 128, 0, stream>>>(
        W1im, planePair, AmH, AmL, W1re, planePair, BmH, BmL, 1.f, W2im, planePair);
    // Wiener multiply (Zf into W1 buffers)
    wiener_filter_kernel<<<(IMGS * NNE) / 256, 256, 0, stream>>>(G, W2re, W2im, W1re, W1im);
    // G3: U^T = (Zf * (A - iB))^T
    gemm_bf16x3_kernel<true, false><<<gg, 128, 0, stream>>>(
        W1re, planePair, AmH, AmL, W1im, planePair, BmH, BmL, 1.f, W2re, planePair);
    gemm_bf16x3_kernel<true, false><<<gg, 128, 0, stream>>>(
        W1im, planePair, AmH, AmL, W1re, planePair, BnH, BnL, 1.f, W2im, planePair);
    // G4: x = clip( (U^T * (A - iB))^T real part / N^2 )  = (W2re*A + W2im*B)^T
    gemm_bf16x3_kernel<true, true><<<gg, 128, 0, stream>>>(
        W2re, planePair, AmH, AmL, W2im, planePair, BmH, BmL, inv_nn, out, (unsigned long long)NNE);
}
